// RTRLDiagonalRNN_64510408786262
// MI455X (gfx1250) — matrix operand order repaired and checked by reading, not compiled, not run on hardware
//
#include <hip/hip_runtime.h>

// ---------------------------------------------------------------------------
// RTRL diagonal RNN:  h_t = lamda ⊙ h_{t-1} + B @ x_t,  h_0 = 0
//
// Phase 0 (if ws allows): split fp32 X, B into bf16 hi/lo pairs once
//                         (bf16x3 split => fp32-class accuracy on bf16 WMMA).
// Phase 1: Y = X @ B^T as a WMMA GEMM, async global->LDS double-buffered
//          staging (ASYNCcnt pipeline). 34.4 GFLOP -> matrix-unit bound.
// Phase 2: per-channel linear scan over T, chunk-parallel (bandwidth trivial;
//          whole working set fits in the 192 MB L2).
// ---------------------------------------------------------------------------

#define LDSS   40      // LDS row stride in ushorts: 32 data + 8 pad
#define TILE   128     // M and N tile per workgroup
#define KSLAB  32      // K per WMMA
#define SCAN_L 64      // timesteps per scan chunk

typedef unsigned short u16;
typedef __attribute__((ext_vector_type(16))) __bf16          v16bf;
typedef __attribute__((ext_vector_type(8)))  __bf16          v8bf;
typedef __attribute__((ext_vector_type(8)))  float           v8f;
typedef __attribute__((ext_vector_type(8)))  unsigned short  us8;
typedef int v4i __attribute__((vector_size(16)));

#if defined(__has_builtin)
# if __has_builtin(__builtin_amdgcn_global_load_async_to_lds_b128)
#  define HAVE_ASYNC_LDS 1
# endif
#endif
#ifndef HAVE_ASYNC_LDS
# define HAVE_ASYNC_LDS 0
#endif

static __device__ __forceinline__ u16 f2bf(float f) {
  union { float f; unsigned int u; } c; c.f = f;
  unsigned int u = c.u;
  unsigned int r = u + 0x7FFFu + ((u >> 16) & 1u);   // round-to-nearest-even
  return (u16)(r >> 16);
}
static __device__ __forceinline__ float bf2f(u16 s) {
  union { unsigned int u; float f; } c; c.u = ((unsigned int)s) << 16;
  return c.f;
}

// Copy 16 bytes global -> LDS. Async (ASYNCcnt) when the builtin exists.
// Builtin signature: (v4i as(1)* gsrc, v4i as(3)* ldst, imm offset, imm cpol).
static __device__ __forceinline__ void stage16B(u16* lds_dst, const u16* g_src) {
#if HAVE_ASYNC_LDS
  __builtin_amdgcn_global_load_async_to_lds_b128(
      (__attribute__((address_space(1))) v4i*)(void*)g_src,
      (__attribute__((address_space(3))) v4i*)lds_dst,
      /*offset=*/0, /*cpol=*/0);
#else
  *(us8*)lds_dst = *(const us8*)g_src;
#endif
}
static __device__ __forceinline__ void async_wait0() {
#if HAVE_ASYNC_LDS
# if __has_builtin(__builtin_amdgcn_s_wait_asynccnt)
  __builtin_amdgcn_s_wait_asynccnt(0);
# else
  asm volatile("s_wait_asynccnt 0" ::: "memory");
# endif
#endif
}

// Assemble a v16bf fragment from two 16-byte LDS reads (ds_load_b128 x2).
static __device__ __forceinline__ v16bf frag_ld(const u16* base,
                                                int row, int o0, int o1) {
  v8bf a = *(const v8bf*)(base + row * LDSS + o0);
  v8bf b = *(const v8bf*)(base + row * LDSS + o1);
  return __builtin_shufflevector(a, b, 0,1,2,3,4,5,6,7,8,9,10,11,12,13,14,15);
}

// ---------------------------------------------------------------------------
// Phase 0: split fp32 -> bf16 hi/lo (x ~= hi + lo), vectorized.
// ---------------------------------------------------------------------------
__global__ __launch_bounds__(256)
void cvt_split(const float4* __restrict__ src, ushort4* __restrict__ hi,
               ushort4* __restrict__ lo, int n4) {
  int i = blockIdx.x * 256 + threadIdx.x;
  if (i >= n4) return;
  float4 v = src[i];
  float f[4] = {v.x, v.y, v.z, v.w};
  u16 h[4], l[4];
  #pragma unroll
  for (int q = 0; q < 4; ++q) {
    h[q] = f2bf(f[q]);
    l[q] = f2bf(f[q] - bf2f(h[q]));
  }
  hi[i] = make_ushort4(h[0], h[1], h[2], h[3]);
  lo[i] = make_ushort4(l[0], l[1], l[2], l[3]);
}

// ---------------------------------------------------------------------------
// Phase 1 (fast path): Y[t,j] = sum_k X[t,k]*B[j,k] from pre-split bf16.
// 256 threads (8 waves) per 128x128 tile; wave w owns 32x64 -> 2x4 wmma tiles.
// Double-buffered LDS; next K-slab staged with async global->LDS copies while
// WMMAs consume the current slab.
// wave32 fragment layouts as loaded here (contiguous k, same for A and B):
//   A (16x32): lane holds row lane&15; lanes<16 -> K 0-15,
//              lanes>=16 -> K 16-31 (two 8-elem runs).
//   B (32x16): lane holds col lane&15; lanes<16 -> K 0-15, >=16 -> K 16-31.
// ---------------------------------------------------------------------------
__global__ __launch_bounds__(256)
void rtrl_gemm_bf16pre(const u16* __restrict__ Xhi, const u16* __restrict__ Xlo,
                       const u16* __restrict__ Bhi, const u16* __restrict__ Blo,
                       float* __restrict__ Y, int H) {
  __shared__ u16 sAhi[2][TILE * LDSS];
  __shared__ u16 sAlo[2][TILE * LDSS];
  __shared__ u16 sBhi[2][TILE * LDSS];
  __shared__ u16 sBlo[2][TILE * LDSS];

  const int tid  = threadIdx.x;
  const int lane = tid & 31;          // wave32
  const int wave = tid >> 5;
  const int j0 = blockIdx.x * TILE;
  const int t0 = blockIdx.y * TILE;

  const int rowBase = (wave & 3) * 32;
  const int colBase = (wave >> 2) * 64;

  v8f zero = {0.f, 0.f, 0.f, 0.f, 0.f, 0.f, 0.f, 0.f};
  v8f acc[2][4];
  #pragma unroll
  for (int mi = 0; mi < 2; ++mi)
    #pragma unroll
    for (int ni = 0; ni < 4; ++ni) acc[mi][ni] = zero;

  const int aOff = (lane < 16) ? 0 : 16;
  const int bOff = (lane < 16) ? 0 : 16;
  const int mrow = lane & 15;

  // Stage one 128x32 bf16 slab of all four operand arrays into buffer `buf`.
  auto stage = [&](int buf, int kk) {
    #pragma unroll
    for (int i = 0; i < 2; ++i) {
      int chunk = tid + i * 256;        // 0..511 : 16B segments of the slab
      int r = chunk >> 2;               // row 0..127
      int c = (chunk & 3) << 3;         // col 0,8,16,24
      size_t ga = (size_t)(t0 + r) * H + kk + c;
      size_t gb = (size_t)(j0 + r) * H + kk + c;
      stage16B(&sAhi[buf][r * LDSS + c], Xhi + ga);
      stage16B(&sAlo[buf][r * LDSS + c], Xlo + ga);
      stage16B(&sBhi[buf][r * LDSS + c], Bhi + gb);
      stage16B(&sBlo[buf][r * LDSS + c], Blo + gb);
    }
  };

  stage(0, 0);
  async_wait0();
  __syncthreads();

  int cur = 0;
  for (int k0 = 0; k0 < H; k0 += KSLAB) {
    int nxt = cur ^ 1;
    if (k0 + KSLAB < H) stage(nxt, k0 + KSLAB);   // overlap copy with math

    v16bf ahi[2], alo[2], bhi[4], blo[4];
    #pragma unroll
    for (int mi = 0; mi < 2; ++mi) {
      int row = rowBase + mi * 16 + mrow;
      ahi[mi] = frag_ld(sAhi[cur], row, aOff, aOff + 8);
      alo[mi] = frag_ld(sAlo[cur], row, aOff, aOff + 8);
    }
    #pragma unroll
    for (int ni = 0; ni < 4; ++ni) {
      int row = colBase + ni * 16 + mrow;
      bhi[ni] = frag_ld(sBhi[cur], row, bOff, bOff + 8);
      blo[ni] = frag_ld(sBlo[cur], row, bOff, bOff + 8);
    }

    #pragma unroll
    for (int mi = 0; mi < 2; ++mi)
      #pragma unroll
      for (int ni = 0; ni < 4; ++ni) {
        acc[mi][ni] = __builtin_amdgcn_wmma_f32_16x16x32_bf16(
            false, ahi[mi], false, bhi[ni], (short)0, acc[mi][ni], false, false);
        acc[mi][ni] = __builtin_amdgcn_wmma_f32_16x16x32_bf16(
            false, ahi[mi], false, blo[ni], (short)0, acc[mi][ni], false, false);
        acc[mi][ni] = __builtin_amdgcn_wmma_f32_16x16x32_bf16(
            false, alo[mi], false, bhi[ni], (short)0, acc[mi][ni], false, false);
      }

    async_wait0();       // our async copies into `nxt` have landed
    __syncthreads();     // everyone done reading `cur`, copies visible
    cur = nxt;
  }

  const int mAdd = (lane < 16) ? 0 : 8;
  #pragma unroll
  for (int mi = 0; mi < 2; ++mi)
    #pragma unroll
    for (int ni = 0; ni < 4; ++ni)
      #pragma unroll
      for (int r = 0; r < 8; ++r) {
        int t = t0 + rowBase + mi * 16 + r + mAdd;
        int j = j0 + colBase + ni * 16 + (lane & 15);
        Y[(size_t)t * H + j] = acc[mi][ni][r];
      }
}

// ---------------------------------------------------------------------------
// Phase 1 (fallback, small d_ws): fused fp32->bf16 split inside the GEMM.
// ---------------------------------------------------------------------------
__global__ __launch_bounds__(256)
void rtrl_gemm_bf16x3(const float* __restrict__ X, const float* __restrict__ Bm,
                      float* __restrict__ Y, int H) {
  __shared__ u16 sAhi[TILE * LDSS];
  __shared__ u16 sAlo[TILE * LDSS];
  __shared__ u16 sBhi[TILE * LDSS];
  __shared__ u16 sBlo[TILE * LDSS];

  const int tid  = threadIdx.x;
  const int lane = tid & 31;
  const int wave = tid >> 5;
  const int j0 = blockIdx.x * TILE;
  const int t0 = blockIdx.y * TILE;

  const int rowBase = (wave & 3) * 32;
  const int colBase = (wave >> 2) * 64;

  v8f zero = {0.f, 0.f, 0.f, 0.f, 0.f, 0.f, 0.f, 0.f};
  v8f acc[2][4];
  #pragma unroll
  for (int mi = 0; mi < 2; ++mi)
    #pragma unroll
    for (int ni = 0; ni < 4; ++ni) acc[mi][ni] = zero;

  const int aOff = (lane < 16) ? 0 : 16;
  const int bOff = (lane < 16) ? 0 : 16;
  const int mrow = lane & 15;

  for (int k0 = 0; k0 < H; k0 += KSLAB) {
    #pragma unroll
    for (int i = 0; i < 4; ++i) {
      int idx = tid + i * 256;
      int r   = idx >> 3;
      int c   = (idx & 7) << 2;
      float4 va = *(const float4*)(X  + (size_t)(t0 + r) * H + k0 + c);
      float4 vb = *(const float4*)(Bm + (size_t)(j0 + r) * H + k0 + c);
      if (k0 + KSLAB < H) {
        __builtin_prefetch(X  + (size_t)(t0 + r) * H + k0 + KSLAB + c, 0, 1);
        __builtin_prefetch(Bm + (size_t)(j0 + r) * H + k0 + KSLAB + c, 0, 1);
      }
      float fa[4] = {va.x, va.y, va.z, va.w};
      float fb[4] = {vb.x, vb.y, vb.z, vb.w};
      #pragma unroll
      for (int q = 0; q < 4; ++q) {
        u16 ha = f2bf(fa[q]);
        u16 hb = f2bf(fb[q]);
        sAhi[r * LDSS + c + q] = ha;
        sAlo[r * LDSS + c + q] = f2bf(fa[q] - bf2f(ha));
        sBhi[r * LDSS + c + q] = hb;
        sBlo[r * LDSS + c + q] = f2bf(fb[q] - bf2f(hb));
      }
    }
    __syncthreads();

    v16bf ahi[2], alo[2], bhi[4], blo[4];
    #pragma unroll
    for (int mi = 0; mi < 2; ++mi) {
      int row = rowBase + mi * 16 + mrow;
      ahi[mi] = frag_ld(sAhi, row, aOff, aOff + 8);
      alo[mi] = frag_ld(sAlo, row, aOff, aOff + 8);
    }
    #pragma unroll
    for (int ni = 0; ni < 4; ++ni) {
      int row = colBase + ni * 16 + mrow;
      bhi[ni] = frag_ld(sBhi, row, bOff, bOff + 8);
      blo[ni] = frag_ld(sBlo, row, bOff, bOff + 8);
    }
    #pragma unroll
    for (int mi = 0; mi < 2; ++mi)
      #pragma unroll
      for (int ni = 0; ni < 4; ++ni) {
        acc[mi][ni] = __builtin_amdgcn_wmma_f32_16x16x32_bf16(
            false, ahi[mi], false, bhi[ni], (short)0, acc[mi][ni], false, false);
        acc[mi][ni] = __builtin_amdgcn_wmma_f32_16x16x32_bf16(
            false, ahi[mi], false, blo[ni], (short)0, acc[mi][ni], false, false);
        acc[mi][ni] = __builtin_amdgcn_wmma_f32_16x16x32_bf16(
            false, alo[mi], false, bhi[ni], (short)0, acc[mi][ni], false, false);
      }
    __syncthreads();
  }

  const int mAdd = (lane < 16) ? 0 : 8;
  #pragma unroll
  for (int mi = 0; mi < 2; ++mi)
    #pragma unroll
    for (int ni = 0; ni < 4; ++ni)
      #pragma unroll
      for (int r = 0; r < 8; ++r) {
        int t = t0 + rowBase + mi * 16 + r + mAdd;
        int j = j0 + colBase + ni * 16 + (lane & 15);
        Y[(size_t)t * H + j] = acc[mi][ni][r];
      }
}

// ---------------------------------------------------------------------------
// Phase 2: chunked linear scan  h_t = lam*h_{t-1} + y_t (exact decomposition)
// ---------------------------------------------------------------------------
__global__ __launch_bounds__(256)
void scan_partial(const float* __restrict__ Y, const float* __restrict__ lam,
                  float* __restrict__ S, int H) {
  int j = blockIdx.x * 256 + threadIdx.x;
  int c = blockIdx.y;
  float l = lam[j];
  float s = 0.f;
  const float* p = Y + (size_t)c * SCAN_L * H + j;
  for (int k = 0; k < SCAN_L; ++k) s = fmaf(l, s, p[(size_t)k * H]);
  S[(size_t)c * H + j] = s;
}

__global__ __launch_bounds__(256)
void scan_carry(const float* __restrict__ S, const float* __restrict__ lam,
                float* __restrict__ carry, int H, int C) {
  int j = blockIdx.x * 256 + threadIdx.x;
  float l = lam[j];
  float lL = 1.f;
  for (int k = 0; k < SCAN_L; ++k) lL *= l;   // lam^L
  float h = 0.f;
  for (int c = 0; c < C; ++c) {
    carry[(size_t)c * H + j] = h;
    h = fmaf(lL, h, S[(size_t)c * H + j]);
  }
}

__global__ __launch_bounds__(256)
void scan_apply(float* __restrict__ Y, const float* __restrict__ lam,
                const float* __restrict__ carry, int H) {
  int j = blockIdx.x * 256 + threadIdx.x;
  int c = blockIdx.y;
  float l = lam[j];
  float h = carry[(size_t)c * H + j];
  float* p = Y + (size_t)c * SCAN_L * H + j;
  for (int k = 0; k < SCAN_L; ++k) {
    h = fmaf(l, h, p[(size_t)k * H]);
    p[(size_t)k * H] = h;
  }
}

// ---------------------------------------------------------------------------
extern "C" void kernel_launch(void* const* d_in, const int* in_sizes, int n_in,
                              void* d_out, int out_size, void* d_ws, size_t ws_size,
                              hipStream_t stream) {
  const float* X   = (const float*)d_in[0];   // [T, H]
  const float* lam = (const float*)d_in[1];   // [H]
  const float* Bm  = (const float*)d_in[2];   // [H, H]
  float* out = (float*)d_out;                 // [T, H]

  const int H  = in_sizes[1];
  const int T  = in_sizes[0] / H;
  const int C  = T / SCAN_L;
  const size_t nX = (size_t)in_sizes[0];
  const size_t nB = (size_t)in_sizes[2];

  float* S     = (float*)d_ws;                // [C, H]
  float* carry = S + (size_t)C * H;           // [C, H]
  const size_t scanBytes = 2ull * (size_t)C * H * sizeof(float);
  const size_t preBytes  = 2ull * (nX + nB) * sizeof(u16);  // hi+lo for X and B

  dim3 g1(H / TILE, T / TILE);

  if (ws_size >= scanBytes + preBytes) {
    // Fast path: pre-split bf16 hi/lo, async-staged WMMA GEMM.
    u16* Xhi = (u16*)((char*)d_ws + scanBytes);
    u16* Xlo = Xhi + nX;
    u16* Bhi = Xlo + nX;
    u16* Blo = Bhi + nB;
    int nx4 = (int)(nX / 4), nb4 = (int)(nB / 4);
    cvt_split<<<(nx4 + 255) / 256, 256, 0, stream>>>(
        (const float4*)X, (ushort4*)Xhi, (ushort4*)Xlo, nx4);
    cvt_split<<<(nb4 + 255) / 256, 256, 0, stream>>>(
        (const float4*)Bm, (ushort4*)Bhi, (ushort4*)Blo, nb4);
    rtrl_gemm_bf16pre<<<g1, 256, 0, stream>>>(Xhi, Xlo, Bhi, Blo, out, H);
  } else {
    // Fallback: fused conversion GEMM (only scan scratch needed).
    rtrl_gemm_bf16x3<<<g1, 256, 0, stream>>>(X, Bm, out, H);
  }

  scan_partial<<<dim3(H / 256, C), 256, 0, stream>>>(out, lam, S, H);
  scan_carry  <<<dim3(H / 256),    256, 0, stream>>>(S, lam, carry, H, C);
  scan_apply  <<<dim3(H / 256, C), 256, 0, stream>>>(out, lam, carry, H);
}
